// DisparityNetworkStage0_22531398435047
// MI455X (gfx1250) — matrix operand order repaired and checked by reading, not compiled, not run on hardware
//
#include <hip/hip_runtime.h>
#include <hip/hip_bf16.h>

// ---------------------------------------------------------------------------
// MI455X (gfx1250) disparity-network stage.  Memory-bound (~0.9 GB traffic vs
// ~20 GFLOP at 23.3 TB/s => ~40us floor): f16 activations, WMMA implicit-GEMM
// convs, LDS halo tiles, wide (b64/b128) LDS & global accesses, and a
// precomputed tap-offset table so per-element address VALU disappears.
// ---------------------------------------------------------------------------

typedef __attribute__((ext_vector_type(16))) _Float16 v16h;
typedef __attribute__((ext_vector_type(8)))  _Float16 v8h;
typedef __attribute__((ext_vector_type(4)))  _Float16 v4h;
typedef __attribute__((ext_vector_type(8)))  float    v8f;

constexpr int Bn = 16, Hn = 128, Wn = 256, Cn = 64, Dn = 12;
constexpr int HOUT = 1024, WOUT = 2048;
constexpr int WT = Wn / 16;                 // w-tiles per row

// Halo tile: 3(h) x 18(w) x 14(d) voxels of 4 f16 channels = 756 voxels,
// plus a 224-voxel zero pad that out-of-range taps are redirected into.
constexpr int TILE_VOX = 3 * 18 * 14;       // 756
constexpr int PAD_VOX  = 16 * 14;           // covers max m*14+d = 223
constexpr int TILE_TOT = TILE_VOX + PAD_VOX;// 980 voxels (7840 B)

// ---------------------------------------------------------------------------
// Kernel 1: cost volume.  cost[b,h,w,d] = sum_c |L[b,h,w,c] - R[b,h,w-d,c]|
// (R zero-padded for w<d).  L row in registers; R rows get 12x reuse out of
// the 192 MB L2.  Output: f16 4-channel padded activation (ch0=cost).
// ---------------------------------------------------------------------------
__global__ __launch_bounds__(256)
void cost_volume_kernel(const float* __restrict__ fl,
                        const float* __restrict__ fr,
                        _Float16* __restrict__ x0) {
  int bh = blockIdx.x;                      // b*H + h
  int w  = threadIdx.x;                     // blockDim.x == Wn == 256
  const float4* lp = (const float4*)(fl + ((size_t)bh * Wn + w) * Cn);
  float4 L[Cn / 4];
#pragma unroll
  for (int c = 0; c < Cn / 4; ++c) L[c] = lp[c];

  for (int d = 0; d < Dn; ++d) {
    float s = 0.f;
    if (w >= d) {
      const float4* rp = (const float4*)(fr + ((size_t)bh * Wn + (w - d)) * Cn);
#pragma unroll
      for (int c = 0; c < Cn / 4; ++c) {
        float4 a = L[c], r = rp[c];
        s += fabsf(a.x - r.x) + fabsf(a.y - r.y) +
             fabsf(a.z - r.z) + fabsf(a.w - r.w);
      }
    } else {
#pragma unroll
      for (int c = 0; c < Cn / 4; ++c) {
        float4 a = L[c];
        s += fabsf(a.x) + fabsf(a.y) + fabsf(a.z) + fabsf(a.w);
      }
    }
    v4h val = {(_Float16)s, (_Float16)0.0f, (_Float16)0.0f, (_Float16)0.0f};
    ((v4h*)x0)[((size_t)bh * Wn + w) * Dn + d] = val;   // one b64 store
  }
}

// ---------------------------------------------------------------------------
// Kernel 2 (template): 3x3x3 conv3d via WMMA implicit GEMM.
//   Per wave: M = 16 w-positions, N = 16 chan slots (COUT valid),
//   K = 27 taps x 4 padded-Cin = 108 -> 4 chunks of 32 (zero-padded exactly).
// Block = 384 threads = 12 waves = 12 disparity planes of one (b,h,w-tile).
// K-mapping tap=K>>2, ci=K&3 makes each tap's 4 channels contiguous, so the
// A gather is pure ds_load_b64 with table-driven addressing; B is ds_load_b128
// from a [n][k]-transposed weight matrix.
// ---------------------------------------------------------------------------
template <int CIN, int COUT, bool RELU, bool FINAL>
__global__ __launch_bounds__(384)
void conv3d_wmma_kernel(const _Float16* __restrict__ x,   // [B,H,W,D,4] f16
                        const float* __restrict__ wgt,     // [3,3,3,CIN,COUT]
                        const float* __restrict__ bias,    // [COUT]
                        _Float16* __restrict__ y,          // [B,H,W,D,4] if !FINAL
                        float* __restrict__ yf) {          // [B,H,W,D]   if FINAL
  __shared__ __align__(16) _Float16 tile[TILE_TOT * 4];  // halo + zero pad
  __shared__ __align__(16) _Float16 Bmat[16 * 128];      // [n][k] weights
  __shared__ int tapOff[32];                             // tap -> voxel offset

  __builtin_prefetch(wgt, 0, 1);               // global_prefetch_b8

  int blk = blockIdx.x;                        // B*H*WT blocks
  int wt  = blk % WT;
  int bh  = blk / WT;
  int h   = bh % Hn;
  int b   = bh / Hn;
  int w0  = wt * 16;
  int tid = threadIdx.x;

  // tap -> voxel offset table; invalid taps point at the zero pad.
  if (tid < 32) {
    int t = tid, off = TILE_VOX;
    if (t < 27) {
      int kh = t / 9, rem = t - kh * 9;
      int kw = rem / 3, kd = rem - kw * 3;
      off = kh * 252 + kw * 14 + kd;
    }
    tapOff[tid] = off;
  }

  // Expand weights transposed: Bmat[n*128 + k] = w[tap][ci][n], zero padded.
  for (int i = tid; i < 16 * 128; i += 384) {
    int n = i >> 7, k = i & 127;
    int tap = k >> 2, ci = k & 3;
    _Float16 v = (_Float16)0.0f;
    if (tap < 27 && ci < CIN && n < COUT)
      v = (_Float16)wgt[(tap * CIN + ci) * COUT + n];
    Bmat[i] = v;
  }

  // Fill halo tile (b64 per voxel) + zero pad region.
  for (int v = tid; v < TILE_TOT; v += 384) {
    v4h val = {};
    if (v < TILE_VOX) {
      int dd = v % 14, t = v / 14;
      int dw = t % 18, dh = t / 18;
      int hh = h + dh - 1;
      int ww = w0 + dw - 1;
      int d  = dd - 1;
      if ((unsigned)hh < (unsigned)Hn && (unsigned)ww < (unsigned)Wn &&
          (unsigned)d < (unsigned)Dn) {
        val = ((const v4h*)x)[(((size_t)b * Hn + hh) * Wn + ww) * Dn + d];
      }
    }
    ((v4h*)tile)[v] = val;
  }
  __syncthreads();

  int wave = tid >> 5;      // 0..11 -> disparity plane
  int lane = tid & 31;
  int d    = wave;
  int n    = lane & 15;     // B column / D column
  int m    = lane & 15;     // A row
  int kb   = lane >> 4;     // K half selector (wave32 A/B layouts)
  int md   = m * 14 + d;    // per-lane voxel offset

  const v4h* tilev = (const v4h*)tile;
  v8f acc = {};
#pragma unroll
  for (int ch = 0; ch < 4; ++ch) {
    // B chunk: 16 consecutive k for column n -> two ds_load_b128.
    const v8h* bp = (const v8h*)(Bmat + n * 128 + ch * 32 + kb * 16);
    v8h bl = bp[0], bh2 = bp[1];
    v16h Bf = __builtin_shufflevector(bl, bh2, 0, 1, 2, 3, 4, 5, 6, 7,
                                      8, 9, 10, 11, 12, 13, 14, 15);
    // A chunk: 4 taps x 4 channels -> four ds_load_b64 via offset table.
    int tb = ch * 8 + kb * 4;
    v4h a0 = tilev[tapOff[tb + 0] + md];
    v4h a1 = tilev[tapOff[tb + 1] + md];
    v4h a2 = tilev[tapOff[tb + 2] + md];
    v4h a3 = tilev[tapOff[tb + 3] + md];
    v8h lo = __builtin_shufflevector(a0, a1, 0, 1, 2, 3, 4, 5, 6, 7);
    v8h hi = __builtin_shufflevector(a2, a3, 0, 1, 2, 3, 4, 5, 6, 7);
    v16h Af = __builtin_shufflevector(lo, hi, 0, 1, 2, 3, 4, 5, 6, 7,
                                      8, 9, 10, 11, 12, 13, 14, 15);
    acc = __builtin_amdgcn_wmma_f32_16x16x32_f16(
        false, Af, false, Bf, (short)0, acc, false, false);
  }

  // D layout: element (M = r + 8*kb, N = lane&15) in acc[r].
  float bv = (n < COUT) ? bias[n] : 0.f;
  int mbase = kb * 8;
#pragma unroll
  for (int r = 0; r < 8; ++r) {
    float v = acc[r] + bv;
    if (RELU) v = fmaxf(v, 0.f);
    int w = w0 + r + mbase;
    if (FINAL) {
      if (n == 0)
        yf[(((size_t)b * Hn + h) * Wn + w) * Dn + d] = v;
    } else {
      if (n < 4) {
        _Float16 hv = (n < COUT) ? (_Float16)v : (_Float16)0.0f;
        y[((((size_t)b * Hn + h) * Wn + w) * Dn + d) * 4 + n] = hv;
      }
    }
  }
}

// ---------------------------------------------------------------------------
// Kernel 3: softmax(-cost) over D + disparity expectation -> pred[b,h,w].
// ---------------------------------------------------------------------------
__global__ __launch_bounds__(256)
void softmax_disp_kernel(const float* __restrict__ cost,
                         float* __restrict__ pred) {
  size_t i = (size_t)blockIdx.x * blockDim.x + threadIdx.x;
  if (i >= (size_t)Bn * Hn * Wn) return;
  const float* c = cost + i * Dn;
  float l[Dn];
  float mx = -1e30f;
#pragma unroll
  for (int d = 0; d < Dn; ++d) { l[d] = -c[d]; mx = fmaxf(mx, l[d]); }
  float se = 0.f, sd = 0.f;
#pragma unroll
  for (int d = 0; d < Dn; ++d) {
    float e = __expf(l[d] - mx);
    se += e;
    sd += e * (float)d;
  }
  pred[i] = sd / se;
}

// ---------------------------------------------------------------------------
// Kernel 4: 8x bilinear upsample (half-pixel centers, edge clamp), matching
// jax.image.resize(..., method='bilinear').
// ---------------------------------------------------------------------------
__global__ __launch_bounds__(256)
void resize_bilinear_kernel(const float* __restrict__ pred,
                            float* __restrict__ out) {
  size_t i = (size_t)blockIdx.x * blockDim.x + threadIdx.x;
  if (i >= (size_t)Bn * HOUT * WOUT) return;
  int ox = (int)(i % WOUT);
  size_t t = i / WOUT;
  int oy = (int)(t % HOUT);
  int b  = (int)(t / HOUT);

  float sy = (oy + 0.5f) * ((float)Hn / HOUT) - 0.5f;
  float sx = (ox + 0.5f) * ((float)Wn / WOUT) - 0.5f;
  sy = fminf(fmaxf(sy, 0.f), (float)(Hn - 1));
  sx = fminf(fmaxf(sx, 0.f), (float)(Wn - 1));
  int y0 = (int)sy, x0 = (int)sx;
  float fy = sy - y0, fx = sx - x0;
  int y1 = min(y0 + 1, Hn - 1);
  int x1 = min(x0 + 1, Wn - 1);

  const float* p = pred + (size_t)b * Hn * Wn;
  float v00 = p[(size_t)y0 * Wn + x0], v01 = p[(size_t)y0 * Wn + x1];
  float v10 = p[(size_t)y1 * Wn + x0], v11 = p[(size_t)y1 * Wn + x1];
  out[i] = (1.f - fy) * ((1.f - fx) * v00 + fx * v01) +
           fy * ((1.f - fx) * v10 + fx * v11);
}

// ---------------------------------------------------------------------------
// Host launcher.  Workspace (~96 MiB ping-pong):
//   bufA, bufB: [B,H,W,D,4] f16 (50,331,648 B each); final f32 cost and pred
//   overlap the free bufB region during/after the last conv (reads bufA only).
// ---------------------------------------------------------------------------
extern "C" void kernel_launch(void* const* d_in, const int* in_sizes, int n_in,
                              void* d_out, int out_size, void* d_ws, size_t ws_size,
                              hipStream_t stream) {
  const float* fl = (const float*)d_in[0];
  const float* fr = (const float*)d_in[1];
  const float* w0 = (const float*)d_in[2];  const float* b0 = (const float*)d_in[3];
  const float* w1 = (const float*)d_in[4];  const float* b1 = (const float*)d_in[5];
  const float* w2 = (const float*)d_in[6];  const float* b2 = (const float*)d_in[7];
  const float* w3 = (const float*)d_in[8];  const float* b3 = (const float*)d_in[9];
  const float* wf = (const float*)d_in[10]; const float* bf = (const float*)d_in[11];
  float* out = (float*)d_out;

  const size_t nvox = (size_t)Bn * Hn * Wn * Dn;         // 6,291,456
  const size_t actBytes = nvox * 4 * sizeof(_Float16);   // 50,331,648
  char* ws = (char*)d_ws;
  _Float16* bufA = (_Float16*)ws;
  _Float16* bufB = (_Float16*)(ws + actBytes);
  float* costf = (float*)bufB;                           // overlaps free bufB
  float* pred  = (float*)(ws + actBytes + nvox * sizeof(float));

  const unsigned NB = Bn * Hn * WT;                      // 32768 conv blocks

  cost_volume_kernel<<<Bn * Hn, 256, 0, stream>>>(fl, fr, bufA);

  conv3d_wmma_kernel<1, 4, true,  false><<<NB, 384, 0, stream>>>(bufA, w0, b0, bufB, nullptr);
  conv3d_wmma_kernel<4, 4, true,  false><<<NB, 384, 0, stream>>>(bufB, w1, b1, bufA, nullptr);
  conv3d_wmma_kernel<4, 4, true,  false><<<NB, 384, 0, stream>>>(bufA, w2, b2, bufB, nullptr);
  conv3d_wmma_kernel<4, 4, true,  false><<<NB, 384, 0, stream>>>(bufB, w3, b3, bufA, nullptr);
  conv3d_wmma_kernel<4, 1, false, true ><<<NB, 384, 0, stream>>>(bufA, wf, bf, nullptr, costf);

  {
    size_t n = (size_t)Bn * Hn * Wn;
    softmax_disp_kernel<<<(unsigned)((n + 255) / 256), 256, 0, stream>>>(costf, pred);
  }
  {
    size_t n = (size_t)Bn * HOUT * WOUT;
    resize_bilinear_kernel<<<(unsigned)((n + 255) / 256), 256, 0, stream>>>(pred, out);
  }
}
